// GatedSingleHeadAttention_67542655696929
// MI455X (gfx1250) — matrix operand order repaired and checked by reading, not compiled, not run on hardware
//
#include <hip/hip_runtime.h>
#include <hip/hip_bf16.h>

// ---------------------------------------------------------------------------
// Differential attention + GLU for MI455X (gfx1250, wave32, WMMA).
// All heavy GEMMs via v_wmma_f32_16x16x32_bf16 with f32 accumulation.
// Staging: batched global b128 loads -> LDS; next stage prefetched once per
// thread via global_prefetch_b8 (hoisted out of the load loop so the load
// batch stays fused).
// ---------------------------------------------------------------------------

typedef __bf16 bf16_t;
typedef __attribute__((ext_vector_type(16))) __bf16 v16bf;
typedef __attribute__((ext_vector_type(8)))  __bf16 v8bf;
typedef __attribute__((ext_vector_type(8)))  float  v8f;

#define B_    4
#define S_    2048
#define E_    1024
#define HD_   512
#define DFF_  2048
#define LAMBDA_INIT 0.8f

// ===========================================================================
// Tiled WMMA GEMM:  C(M,N) = A(M,K) * B(N,K)^T   (both row-major bf16)
// Block tile 128x128x64, 256 threads = 8 waves, wave tile 32x64 (2x4 WMMA).
// OUT_MODE: 0 = f32 row-major, 1 = bf16 row-major, 2 = bf16 transposed (N,M)
// ===========================================================================
template<int OUT_MODE>
__global__ __launch_bounds__(256)
void gemm_nt_wmma_bf16(const bf16_t* __restrict__ A, int lda,
                       const bf16_t* __restrict__ Bm, int ldb,
                       void* __restrict__ C, int ldc, int K)
{
    constexpr int BM = 128, BN = 128, BK = 64;
    constexpr int ST = BK + 8;                 // padded LDS stride (keeps 16B align)
    __shared__ __align__(16) bf16_t As[BM * ST];
    __shared__ __align__(16) bf16_t Bs[BN * ST];

    const int tid  = threadIdx.x;
    const int lane = tid & 31;
    const int wid  = tid >> 5;
    const int wm   = wid & 3;                  // wave row slab   (4 x 32 = 128)
    const int wn   = wid >> 2;                 // wave col slab   (2 x 64 = 128)
    const int bm   = blockIdx.y * BM;
    const int bn   = blockIdx.x * BN;

    v8f acc[2][4];
    const v8f vzero = {0.f, 0.f, 0.f, 0.f, 0.f, 0.f, 0.f, 0.f};
    #pragma unroll
    for (int i = 0; i < 2; ++i)
        #pragma unroll
        for (int j = 0; j < 4; ++j) acc[i][j] = vzero;

    const int rowsel = lane & 15;
    const int aoff   = (lane < 16) ? 0 : 8;    // A: K chunk select per half-wave
    const int boff   = (lane < 16) ? 0 : 8;    // B: K chunk select per half-wave

    // Per-thread prefetch target: 256 threads cover 128 A rows + 128 B rows.
    const int  pfr       = tid >> 1;
    const bool pfb       = (tid & 1) != 0;
    const bf16_t* pfbase = pfb ? &Bm[(size_t)(bn + pfr) * ldb]
                               : &A [(size_t)(bm + pfr) * lda];

    for (int kt = 0; kt < K; kt += BK) {
        // ---- speculative prefetch of next K-stage (global_prefetch_b8) ----
        if (kt + BK < K)
            __builtin_prefetch(pfbase + kt + BK, 0, 3);

        // ---- stage 128x64 A tile and 128x64 B tile into LDS ----
        // Load all chunks into registers first so the 8 global b128 loads are
        // in flight concurrently, then drain them into LDS.
        uint4 ra[4], rb[4];
        #pragma unroll
        for (int it = 0; it < 4; ++it) {
            const int idx = tid + it * 256;       // 0..1023
            const int r   = idx >> 3;             // 0..127
            const int c8  = (idx & 7) << 3;       // 0,8,..,56
            ra[it] = *(const uint4*)&A [(size_t)(bm + r) * lda + kt + c8];
            rb[it] = *(const uint4*)&Bm[(size_t)(bn + r) * ldb + kt + c8];
        }
        #pragma unroll
        for (int it = 0; it < 4; ++it) {
            const int idx = tid + it * 256;
            const int r   = idx >> 3;
            const int c8  = (idx & 7) << 3;
            *(uint4*)&As[r * ST + c8] = ra[it];
            *(uint4*)&Bs[r * ST + c8] = rb[it];
        }
        __syncthreads();

        // ---- 2 WMMA K-steps of 32 per stage ----
        #pragma unroll
        for (int ks = 0; ks < BK; ks += 32) {
            v16bf afr[2], bfr[4];
            #pragma unroll
            for (int mi = 0; mi < 2; ++mi) {
                const bf16_t* p = &As[(wm * 32 + mi * 16 + rowsel) * ST + ks + aoff];
                union { v16bf v; v8bf h[2]; } u;
                u.h[0] = *(const v8bf*)(p);        // K chunk [ks+aoff,    +8)
                u.h[1] = *(const v8bf*)(p + 16);   // K chunk [ks+aoff+16, +8)
                afr[mi] = u.v;
            }
            #pragma unroll
            for (int ni = 0; ni < 4; ++ni) {
                const bf16_t* p = &Bs[(wn * 64 + ni * 16 + rowsel) * ST + ks + boff];
                union { v16bf v; v8bf h[2]; } u;
                u.h[0] = *(const v8bf*)(p);        // K chunk [ks+boff,    +8)
                u.h[1] = *(const v8bf*)(p + 16);   // K chunk [ks+boff+16, +8)
                bfr[ni] = u.v;
            }
            #pragma unroll
            for (int mi = 0; mi < 2; ++mi)
                #pragma unroll
                for (int ni = 0; ni < 4; ++ni)
                    acc[mi][ni] = __builtin_amdgcn_wmma_f32_16x16x32_bf16(
                        false, afr[mi], false, bfr[ni],
                        (short)0, acc[mi][ni], false, false);
        }
        __syncthreads();
    }

    // ---- epilogue: C/D layout = lane%16 -> N, (lane/16)*8 + vgpr -> M ----
    const int nloc = lane & 15;
    const int mh   = (lane >> 4) * 8;
    #pragma unroll
    for (int mi = 0; mi < 2; ++mi) {
        #pragma unroll
        for (int ni = 0; ni < 4; ++ni) {
            const int gm = bm + wm * 32 + mi * 16 + mh;
            const int gn = bn + wn * 64 + ni * 16 + nloc;
            if constexpr (OUT_MODE == 0) {
                float* out = (float*)C;
                #pragma unroll
                for (int i = 0; i < 8; ++i)
                    out[(size_t)(gm + i) * ldc + gn] = acc[mi][ni][i];
            } else if constexpr (OUT_MODE == 1) {
                bf16_t* out = (bf16_t*)C;
                #pragma unroll
                for (int i = 0; i < 8; ++i)
                    out[(size_t)(gm + i) * ldc + gn] = (bf16_t)acc[mi][ni][i];
            } else {
                bf16_t* out = (bf16_t*)C;          // transposed: out[n*ldc + m]
                v8bf pk;
                #pragma unroll
                for (int i = 0; i < 8; ++i) pk[i] = (bf16_t)acc[mi][ni][i];
                *(v8bf*)&out[(size_t)gn * ldc + gm] = pk;  // contiguous 16B
            }
        }
    }
}

// ===========================================================================
// Elementwise / reduction helpers
// ===========================================================================
__global__ __launch_bounds__(256)
void cvt_f32_bf16_kernel(const float* __restrict__ in, bf16_t* __restrict__ out,
                         size_t n)
{
    size_t i = (size_t)blockIdx.x * 256 + threadIdx.x;
    if (i < n) out[i] = (bf16_t)in[i];
}

// in: f32 (R,C) row-major  ->  out: bf16 (C,R) row-major
__global__ __launch_bounds__(256)
void cvt_t_f32_bf16_kernel(const float* __restrict__ in, bf16_t* __restrict__ out,
                           int R, int C)
{
    size_t i = (size_t)blockIdx.x * 256 + threadIdx.x;
    if (i >= (size_t)R * C) return;
    const int r = (int)(i / C), c = (int)(i % C);
    out[(size_t)c * R + r] = (bf16_t)in[i];
}

__global__ __launch_bounds__(256)
void lambda_kernel(const float* __restrict__ lq1, const float* __restrict__ lk1,
                   const float* __restrict__ lq2, const float* __restrict__ lk2,
                   float* __restrict__ lam_out)
{
    __shared__ float r1[256], r2[256];
    const int t = threadIdx.x;
    float a = 0.f, b = 0.f;
    for (int i = t; i < HD_; i += 256) { a += lq1[i] * lk1[i]; b += lq2[i] * lk2[i]; }
    r1[t] = a; r2[t] = b;
    __syncthreads();
    for (int s = 128; s > 0; s >>= 1) {
        if (t < s) { r1[t] += r1[t + s]; r2[t] += r2[t + s]; }
        __syncthreads();
    }
    if (t == 0) *lam_out = __expf(r1[0]) - __expf(r2[0]) + LAMBDA_INIT;
}

__device__ __forceinline__ float block_red_max(float v, float* red, int tid)
{
    #pragma unroll
    for (int off = 16; off > 0; off >>= 1) v = fmaxf(v, __shfl_xor(v, off, 32));
    if ((tid & 31) == 0) red[tid >> 5] = v;
    __syncthreads();
    float r = red[0];
    #pragma unroll
    for (int i = 1; i < 8; ++i) r = fmaxf(r, red[i]);
    __syncthreads();
    return r;
}

__device__ __forceinline__ float block_red_sum(float v, float* red, int tid)
{
    #pragma unroll
    for (int off = 16; off > 0; off >>= 1) v += __shfl_xor(v, off, 32);
    if ((tid & 31) == 0) red[tid >> 5] = v;
    __syncthreads();
    float r = 0.f;
    #pragma unroll
    for (int i = 0; i < 8; ++i) r += red[i];
    __syncthreads();
    return r;
}

// One block per query row: ghostmax over both heads, write diff weights (bf16).
__global__ __launch_bounds__(256)
void softmax_diff_kernel(const float* __restrict__ s0, const float* __restrict__ s1,
                         const float* __restrict__ amask,   // (S,) for this batch
                         const float* __restrict__ lam_p,
                         bf16_t* __restrict__ diff)
{
    constexpr int VPT = S_ / 256;                            // 8
    __shared__ float red[8];
    const int q   = blockIdx.x;
    const int tid = threadIdx.x;
    const float scale = 0.044194173824159216f;               // 1/sqrt(512)

    float x0[VPT], x1[VPT];
    #pragma unroll
    for (int j = 0; j < VPT; ++j) {
        const int k = tid + j * 256;
        const float cm  = (k > q) ? -__builtin_inff() : 0.0f;
        const float pad = (1.0f - amask[k]) * -1e12f;
        const size_t o  = (size_t)q * S_ + k;
        x0[j] = s0[o] * scale + cm + pad;
        x1[j] = s1[o] * scale + cm + pad;
    }
    float m0 = 0.f, m1 = 0.f;                                // ghost zero logit
    #pragma unroll
    for (int j = 0; j < VPT; ++j) { m0 = fmaxf(m0, x0[j]); m1 = fmaxf(m1, x1[j]); }
    m0 = block_red_max(m0, red, tid);
    m1 = block_red_max(m1, red, tid);

    float e0[VPT], e1[VPT], a0 = 0.f, a1 = 0.f;
    #pragma unroll
    for (int j = 0; j < VPT; ++j) {
        e0[j] = __expf(x0[j] - m0); a0 += e0[j];
        e1[j] = __expf(x1[j] - m1); a1 += e1[j];
    }
    a0 = block_red_sum(a0, red, tid);
    a1 = block_red_sum(a1, red, tid);
    const float d0 = 1.0f / (a0 + __expf(-m0));
    const float d1 = 1.0f / (a1 + __expf(-m1));
    const float lam = *lam_p;
    #pragma unroll
    for (int j = 0; j < VPT; ++j) {
        const int k = tid + j * 256;
        diff[(size_t)q * S_ + k] = (bf16_t)(e0[j] * d0 - lam * (e1[j] * d1));
    }
}

// In-place RMSNorm + (1-lambda_init) scaling, one block per row.
__global__ __launch_bounds__(256)
void rmsnorm_kernel(float* __restrict__ x, const float* __restrict__ w)
{
    __shared__ float red[8];
    const int row = blockIdx.x;
    const int tid = threadIdx.x;
    float v[4]; float ss = 0.f;
    #pragma unroll
    for (int j = 0; j < 4; ++j) {
        v[j] = x[(size_t)row * E_ + tid + j * 256];
        ss += v[j] * v[j];
    }
    ss = block_red_sum(ss, red, tid);
    const float inv = rsqrtf(ss * (1.0f / E_) + 1e-6f) * (1.0f - LAMBDA_INIT);
    #pragma unroll
    for (int j = 0; j < 4; ++j) {
        const int e = tid + j * 256;
        x[(size_t)row * E_ + e] = v[j] * inv * w[e];
    }
}

// h (bf16, rows x 2*DFF) -> act = silu(a)*g (bf16, rows x DFF)
__global__ __launch_bounds__(256)
void glu_act_kernel(const bf16_t* __restrict__ h, bf16_t* __restrict__ act, size_t n)
{
    size_t i = (size_t)blockIdx.x * 256 + threadIdx.x;
    if (i >= n) return;
    const size_t r = i / DFF_, c = i % DFF_;
    const float a = (float)h[r * (2 * DFF_) + c];
    const float g = (float)h[r * (2 * DFF_) + DFF_ + c];
    const float s = a / (1.0f + __expf(-a));
    act[i] = (bf16_t)(s * g);
}

// gated = bf16( attn * sigmoid(logits) )
__global__ __launch_bounds__(256)
void gate_mul_kernel(const float* __restrict__ attn, const float* __restrict__ logits,
                     bf16_t* __restrict__ gated, size_t n)
{
    size_t i = (size_t)blockIdx.x * 256 + threadIdx.x;
    if (i >= n) return;
    const float gate = 1.0f / (1.0f + __expf(-logits[i]));
    gated[i] = (bf16_t)(attn[i] * gate);
}

// ===========================================================================
// Host-side orchestration
// ===========================================================================
static inline void launch_gemm(int mode,
                               const bf16_t* A, int lda, const bf16_t* Bm, int ldb,
                               void* C, int ldc, int M, int N, int K,
                               hipStream_t s)
{
    dim3 g(N / 128, M / 128), b(256);
    if (mode == 0)      gemm_nt_wmma_bf16<0><<<g, b, 0, s>>>(A, lda, Bm, ldb, C, ldc, K);
    else if (mode == 1) gemm_nt_wmma_bf16<1><<<g, b, 0, s>>>(A, lda, Bm, ldb, C, ldc, K);
    else                gemm_nt_wmma_bf16<2><<<g, b, 0, s>>>(A, lda, Bm, ldb, C, ldc, K);
}

extern "C" void kernel_launch(void* const* d_in, const int* in_sizes, int n_in,
                              void* d_out, int out_size, void* d_ws, size_t ws_size,
                              hipStream_t stream)
{
    (void)in_sizes; (void)n_in; (void)out_size; (void)ws_size;
    const float* x    = (const float*)d_in[0];
    const float* am   = (const float*)d_in[1];
    const float* Wq   = (const float*)d_in[2];
    const float* Wk   = (const float*)d_in[3];
    const float* Wv   = (const float*)d_in[4];
    const float* Wo   = (const float*)d_in[5];
    const float* lq1  = (const float*)d_in[6];
    const float* lk1  = (const float*)d_in[7];
    const float* lq2  = (const float*)d_in[8];
    const float* lk2  = (const float*)d_in[9];
    const float* rmsw = (const float*)d_in[10];
    const float* gup  = (const float*)d_in[11];
    const float* gdn  = (const float*)d_in[12];
    float* out = (float*)d_out;

    // ---- workspace carve-up ----
    char* w = (char*)d_ws;
    auto carve = [&](size_t bytes) -> char* {
        char* p = w; w += (bytes + 255) & ~(size_t)255; return p;
    };
    const size_t BSE = (size_t)B_ * S_ * E_;                 // 8,388,608
    bf16_t* xb    = (bf16_t*)carve(BSE * 2);
    bf16_t* wq_t  = (bf16_t*)carve((size_t)E_ * E_ * 2);
    bf16_t* wk_t  = (bf16_t*)carve((size_t)E_ * E_ * 2);
    bf16_t* wv_t  = (bf16_t*)carve((size_t)E_ * E_ * 2);
    bf16_t* wo_t  = (bf16_t*)carve((size_t)E_ * E_ * 2);
    bf16_t* gup_t = (bf16_t*)carve((size_t)E_ * 2 * DFF_ * 2);
    bf16_t* gdn_t = (bf16_t*)carve((size_t)DFF_ * E_ * 2);
    bf16_t* qb    = (bf16_t*)carve(BSE * 2);
    bf16_t* kb    = (bf16_t*)carve(BSE * 2);
    bf16_t* vt    = (bf16_t*)carve(BSE * 2);                 // (E, B*S) transposed
    bf16_t* hb    = (bf16_t*)carve((size_t)B_ * S_ * 2 * DFF_ * 2);
    bf16_t* actb  = (bf16_t*)carve((size_t)B_ * S_ * DFF_ * 2);
    float*  sc0   = (float*)carve((size_t)S_ * S_ * 4);      // reused per batch
    float*  sc1   = (float*)carve((size_t)S_ * S_ * 4);
    bf16_t* dw    = (bf16_t*)carve((size_t)S_ * S_ * 2);
    float*  attn  = (float*)carve(BSE * 4);
    float*  glog  = (float*)carve(BSE * 4);
    bf16_t* gated = (bf16_t*)carve(BSE * 2);
    float*  lam   = (float*)carve(256);

    const int M  = B_ * S_;                                  // 8192
    dim3 blk(256);

    // 1) precision conversion (inputs + transposed weights)
    cvt_f32_bf16_kernel<<<dim3((unsigned)(BSE / 256)), blk, 0, stream>>>(x, xb, BSE);
    cvt_t_f32_bf16_kernel<<<dim3((unsigned)((size_t)E_ * E_ / 256)), blk, 0, stream>>>(Wq, wq_t, E_, E_);
    cvt_t_f32_bf16_kernel<<<dim3((unsigned)((size_t)E_ * E_ / 256)), blk, 0, stream>>>(Wk, wk_t, E_, E_);
    cvt_t_f32_bf16_kernel<<<dim3((unsigned)((size_t)E_ * E_ / 256)), blk, 0, stream>>>(Wv, wv_t, E_, E_);
    cvt_t_f32_bf16_kernel<<<dim3((unsigned)((size_t)E_ * E_ / 256)), blk, 0, stream>>>(Wo, wo_t, E_, E_);
    cvt_t_f32_bf16_kernel<<<dim3((unsigned)((size_t)E_ * 2 * DFF_ / 256)), blk, 0, stream>>>(gup, gup_t, E_, 2 * DFF_);
    cvt_t_f32_bf16_kernel<<<dim3((unsigned)((size_t)DFF_ * E_ / 256)), blk, 0, stream>>>(gdn, gdn_t, DFF_, E_);

    // 2) projections (all WMMA)
    launch_gemm(1, xb, E_, wq_t, E_, qb, E_, M, E_, E_, stream);           // Q  (bf16)
    launch_gemm(1, xb, E_, wk_t, E_, kb, E_, M, E_, E_, stream);           // K  (bf16)
    launch_gemm(2, xb, E_, wv_t, E_, vt, M,  M, E_, E_, stream);           // V^T (E, B*S)
    launch_gemm(1, xb, E_, gup_t, E_, hb, 2 * DFF_, M, 2 * DFF_, E_, stream); // GLU up

    // 3) lambda scalar
    lambda_kernel<<<dim3(1), blk, 0, stream>>>(lq1, lk1, lq2, lk2, lam);

    // 4) per-batch differential attention
    for (int b = 0; b < B_; ++b) {
        const bf16_t* qb0 = qb + (size_t)b * S_ * E_;
        const bf16_t* kb0 = kb + (size_t)b * S_ * E_;
        launch_gemm(0, qb0,        E_, kb0,        E_, sc0, S_, S_, S_, HD_, stream);
        launch_gemm(0, qb0 + HD_,  E_, kb0 + HD_,  E_, sc1, S_, S_, S_, HD_, stream);
        softmax_diff_kernel<<<dim3(S_), blk, 0, stream>>>(sc0, sc1, am + (size_t)b * S_, lam, dw);
        launch_gemm(0, dw, S_, vt + (size_t)b * S_, M,
                    attn + (size_t)b * S_ * E_, E_, S_, E_, S_, stream);   // diff_w @ V
    }

    // 5) RMSNorm + (1-lambda) scale, in place
    rmsnorm_kernel<<<dim3(M), blk, 0, stream>>>(attn, rmsw);

    // 6) GLU gate path
    const size_t nact = (size_t)B_ * S_ * DFF_;
    glu_act_kernel<<<dim3((unsigned)(nact / 256)), blk, 0, stream>>>(hb, actb, nact);
    launch_gemm(0, actb, DFF_, gdn_t, DFF_, glog, E_, M, E_, DFF_, stream); // gate logits
    gate_mul_kernel<<<dim3((unsigned)(BSE / 256)), blk, 0, stream>>>(attn, glog, gated, BSE);

    // 7) output projection straight into d_out (f32)
    launch_gemm(0, gated, E_, wo_t, E_, out, E_, M, E_, E_, stream);
}
